// Lift3DFusionNormalizedGraspNet_16733192585550
// MI455X (gfx1250) — matrix operand order repaired and checked by reading, not compiled, not run on hardware
//
#include <hip/hip_runtime.h>
#include <hip/hip_bf16.h>

typedef _Float16 h16;
typedef __attribute__((ext_vector_type(16))) _Float16 v16h;
typedef __attribute__((ext_vector_type(8)))  _Float16 v8h;
typedef __attribute__((ext_vector_type(8)))  float    v8f;

union V16H { v16h v; v8h p[2]; };

static constexpr int BB = 16;    // batch
static constexpr int SS = 1024;  // seeds
static constexpr int MM = 4096;  // points
static constexpr int CC = 768;   // channels in
static constexpr int DD = 256;   // proj out
static constexpr int FLD = 512;  // fused / hidden width

// ---------------------------------------------------------------------------
// WMMA fragment loaders. Memory layout: row-major [rows][ld] with K contiguous.
// A (16x32 f16, ISA 7.12.2): lane L: m=L&15, half=L>>4.
//   VGPR v: k = (v>>2)*16 + half*8 + (v&3)*2  ->  v0..3 = 8 halfs @ kb+half*8,
//                                                 v4..7 = 8 halfs @ kb+16+half*8
// B (32x16 f16): lane L: n=L&15, half=L>>4.
//   VGPR v: k = (v>>2)*16 + half*8 + (v&3)*2  ->  v0..3 @ kb+half*8, v4..7 @ kb+16+half*8
// ---------------------------------------------------------------------------
__device__ __forceinline__ v16h load_frag_a(const h16* __restrict__ base, int ld,
                                            int kb, int lane) {
  const int m = lane & 15, half = lane >> 4;
  const h16* p = base + (size_t)m * ld + kb + half * 8;
  V16H u;
  u.p[0] = *(const v8h*)(p);
  u.p[1] = *(const v8h*)(p + 16);
  return u.v;
}

__device__ __forceinline__ v16h load_frag_b(const h16* __restrict__ base, int ld,
                                            int kb, int lane) {
  const int n = lane & 15, half = lane >> 4;
  const h16* p = base + (size_t)n * ld + kb + half * 8;
  V16H u;
  u.p[0] = *(const v8h*)(p);
  u.p[1] = *(const v8h*)(p + 16);
  return u.v;
}

// ---------------------------------------------------------------------------
// Elementwise f32 -> f16 (weights)
// ---------------------------------------------------------------------------
__global__ void convert_f32_f16(const float* __restrict__ in, h16* __restrict__ out,
                                int n) {
  int i = blockIdx.x * 256 + threadIdx.x;
  if (i < n) out[i] = (h16)in[i];
}

// ---------------------------------------------------------------------------
// Fused: NN argmin over M, gather f_last column, LayerNorm over C, emit f16
// One 256-thread block per (b, s).  Xln layout: [B][S][C] (C contiguous).
// ---------------------------------------------------------------------------
__global__ void nn_gather_ln(const float* __restrict__ xyz,   // [B,S,3]
                             const float* __restrict__ pl,    // [B,M,3]
                             const float* __restrict__ fl,    // [B,C,M]
                             const float* __restrict__ gamma, // [C]
                             const float* __restrict__ beta,  // [C]
                             h16* __restrict__ Xln)           // [B,S,C]
{
  const int s = blockIdx.x, b = blockIdx.y, t = threadIdx.x;
  __shared__ float sd[256];
  __shared__ float ss[256];
  __shared__ int   si[256];

  const float* q = xyz + ((size_t)b * SS + s) * 3;
  const float q0 = q[0], q1 = q[1], q2 = q[2];

  // --- phase 1: argmin_m |q - p|^2 ---
  float best = 3.4e38f;
  int   bidx = 0;
  const float* pb = pl + (size_t)b * MM * 3;
  for (int m = t; m < MM; m += 256) {
    const float* pp = pb + (size_t)m * 3;
    const float dx = q0 - pp[0], dy = q1 - pp[1], dz = q2 - pp[2];
    const float d = dx * dx + dy * dy + dz * dz;
    if (d < best) { best = d; bidx = m; }
  }
  sd[t] = best; si[t] = bidx;
  __syncthreads();
  for (int off = 128; off > 0; off >>= 1) {
    if (t < off) {
      const float dv = sd[t + off];
      const int   iv = si[t + off];
      if (dv < sd[t] || (dv == sd[t] && iv < si[t])) { sd[t] = dv; si[t] = iv; }
    }
    __syncthreads();
  }
  const int idx = si[0];

  // --- phase 2: gather (3 channels/thread) + LayerNorm over C=768 ---
  const float* fb = fl + (size_t)b * CC * MM + idx;
  const float g0 = fb[(size_t)(t)       * MM];
  const float g1 = fb[(size_t)(t + 256) * MM];
  const float g2 = fb[(size_t)(t + 512) * MM];
  __syncthreads();  // si[0] consumed; safe to reuse LDS
  sd[t] = g0 + g1 + g2;
  ss[t] = g0 * g0 + g1 * g1 + g2 * g2;
  __syncthreads();
  for (int off = 128; off > 0; off >>= 1) {
    if (t < off) { sd[t] += sd[t + off]; ss[t] += ss[t + off]; }
    __syncthreads();
  }
  const float mu  = sd[0] * (1.0f / CC);
  const float var = ss[0] * (1.0f / CC) - mu * mu;
  const float inv = rsqrtf(var + 1e-5f);

  h16* xo = Xln + ((size_t)b * SS + s) * CC;
  xo[t]       = (h16)((g0 - mu) * inv * gamma[t]       + beta[t]);
  xo[t + 256] = (h16)((g1 - mu) * inv * gamma[t + 256] + beta[t + 256]);
  xo[t + 512] = (h16)((g2 - mu) * inv * gamma[t + 512] + beta[t + 512]);
}

// ---------------------------------------------------------------------------
// seed_features [B,D,S] f32 -> fused[b][s][0..D-1] f16 (transpose + convert)
// ---------------------------------------------------------------------------
__global__ void seed_transpose(const float* __restrict__ sf, h16* __restrict__ fused) {
  const int s = blockIdx.x * 256 + threadIdx.x;
  const int d = blockIdx.y, b = blockIdx.z;
  fused[((size_t)b * SS + s) * FLD + d] = (h16)sf[((size_t)b * DD + d) * SS + s];
}

// ---------------------------------------------------------------------------
// WMMA GEMM:  Y[o,s] = act( sum_k W[o,k] * X[s,k] + bias[o] )
//   W: [O,K] f16 row-major.  X: per-batch [S,K] f16 row-major.
//   OUT_HALF: write f16 into [B][S][out_ld] at column out_col_off (fused/h).
//   else:     write f32 into [B][O][S] (final output layout).
// Block = 256 threads = 8 waves arranged 2(o) x 4(s) -> 64(o) x 128(s) tile.
// Each wave computes a 32x32 output tile: 2 A-frags x 2 B-frags -> 4 WMMAs
// per K-step (2 b128 loads per wmma; 4 independent accumulator chains).
// ---------------------------------------------------------------------------
template <bool RELU, bool OUT_HALF>
__global__ void gemm_wmma(const h16* __restrict__ W, const h16* __restrict__ X,
                          const float* __restrict__ bias, void* __restrict__ Yv,
                          int O, int K, int out_col_off, int out_ld) {
  const int lane = threadIdx.x & 31;
  const int w    = threadIdx.x >> 5;
  const int b    = blockIdx.z;
  const int o0   = blockIdx.y * 64 + (w & 1) * 32;
  const int s0   = blockIdx.x * 128 + (w >> 1) * 32;

  const h16* Wa0 = W + (size_t)o0 * K;
  const h16* Wa1 = W + (size_t)(o0 + 16) * K;
  const h16* Xb0 = X + (size_t)b * SS * K + (size_t)s0 * K;
  const h16* Xb1 = Xb0 + (size_t)16 * K;

  v8f acc00 = {}, acc01 = {}, acc10 = {}, acc11 = {};
#pragma unroll 2
  for (int kb = 0; kb < K; kb += 32) {
    const v16h a0 = load_frag_a(Wa0, K, kb, lane);
    const v16h a1 = load_frag_a(Wa1, K, kb, lane);
    const v16h b0 = load_frag_b(Xb0, K, kb, lane);
    const v16h b1 = load_frag_b(Xb1, K, kb, lane);
    acc00 = __builtin_amdgcn_wmma_f32_16x16x32_f16(false, a0, false, b0,
                                                   (short)0, acc00, false, false);
    acc01 = __builtin_amdgcn_wmma_f32_16x16x32_f16(false, a0, false, b1,
                                                   (short)0, acc01, false, false);
    acc10 = __builtin_amdgcn_wmma_f32_16x16x32_f16(false, a1, false, b0,
                                                   (short)0, acc10, false, false);
    acc11 = __builtin_amdgcn_wmma_f32_16x16x32_f16(false, a1, false, b1,
                                                   (short)0, acc11, false, false);
  }

  const int half = lane >> 4, n = lane & 15;

#pragma unroll
  for (int i = 0; i < 2; ++i) {        // o sub-tile
    const int orow = o0 + i * 16 + half * 8;   // element r -> output row orow + r
    float bb[8];
    *(float4*)&bb[0] = *(const float4*)(bias + orow);
    *(float4*)&bb[4] = *(const float4*)(bias + orow + 4);
#pragma unroll
    for (int j = 0; j < 2; ++j) {      // s sub-tile
      const v8f acc = (i == 0) ? (j == 0 ? acc00 : acc01)
                               : (j == 0 ? acc10 : acc11);
      const int srow = s0 + j * 16 + n;
      if (OUT_HALF) {
        h16* Y = (h16*)Yv + ((size_t)b * SS + srow) * out_ld + out_col_off + orow;
        v8h o;
#pragma unroll
        for (int r = 0; r < 8; ++r) {
          float v = acc[r] + bb[r];
          if (RELU) v = fmaxf(v, 0.0f);
          o[r] = (h16)v;
        }
        *(v8h*)Y = o;
      } else {
        float* Y = (float*)Yv + ((size_t)b * O + orow) * SS + srow;
#pragma unroll
        for (int r = 0; r < 8; ++r) {
          float v = acc[r] + bb[r];
          if (RELU) v = fmaxf(v, 0.0f);
          Y[(size_t)r * SS] = v;
        }
      }
    }
  }
}

// ---------------------------------------------------------------------------
extern "C" void kernel_launch(void* const* d_in, const int* in_sizes, int n_in,
                              void* d_out, int out_size, void* d_ws, size_t ws_size,
                              hipStream_t stream) {
  const float* xyz   = (const float*)d_in[0];   // [B,S,3]
  const float* plast = (const float*)d_in[1];   // [B,M,3]
  const float* flast = (const float*)d_in[2];   // [B,C,M]
  const float* seedf = (const float*)d_in[3];   // [B,D,S]
  const float* gamma = (const float*)d_in[4];   // [C]
  const float* beta  = (const float*)d_in[5];   // [C]
  const float* projw = (const float*)d_in[6];   // [256,768]
  const float* projb = (const float*)d_in[7];   // [256]
  const float* w1    = (const float*)d_in[8];   // [512,512]
  const float* b1    = (const float*)d_in[9];   // [512]
  const float* w2    = (const float*)d_in[10];  // [256,512]
  const float* b2    = (const float*)d_in[11];  // [256]

  // Workspace carve-up (~57 MB total)
  char* ws = (char*)d_ws;
  h16* Xln   = (h16*)ws;                       ws += (size_t)BB * SS * CC * 2;   // 25.2 MB
  h16* fused = (h16*)ws;                       ws += (size_t)BB * SS * FLD * 2;  // 16.8 MB
  h16* hbuf  = (h16*)ws;                       ws += (size_t)BB * SS * FLD * 2;  // 16.8 MB
  h16* wp_h  = (h16*)ws;                       ws += (size_t)DD * CC * 2;
  h16* w1_h  = (h16*)ws;                       ws += (size_t)FLD * FLD * 2;
  h16* w2_h  = (h16*)ws;                       ws += (size_t)DD * FLD * 2;

  // 1) weights -> f16
  convert_f32_f16<<<(DD * CC + 255) / 256, 256, 0, stream>>>(projw, wp_h, DD * CC);
  convert_f32_f16<<<(FLD * FLD + 255) / 256, 256, 0, stream>>>(w1, w1_h, FLD * FLD);
  convert_f32_f16<<<(DD * FLD + 255) / 256, 256, 0, stream>>>(w2, w2_h, DD * FLD);

  // 2) NN argmin + gather + LayerNorm -> Xln [B,S,C] f16
  nn_gather_ln<<<dim3(SS, BB), 256, 0, stream>>>(xyz, plast, flast, gamma, beta, Xln);

  // 3) seed_features -> fused cols [0,256)
  seed_transpose<<<dim3(SS / 256, DD, BB), 256, 0, stream>>>(seedf, fused);

  // 4) proj GEMM (+bias, ReLU) -> fused cols [256,512)
  gemm_wmma<true, true><<<dim3(SS / 128, DD / 64, BB), 256, 0, stream>>>(
      wp_h, Xln, projb, fused, DD, CC, DD, FLD);

  // 5) mlp1 GEMM (+bias, ReLU) -> hbuf [B,S,512]
  gemm_wmma<true, true><<<dim3(SS / 128, FLD / 64, BB), 256, 0, stream>>>(
      w1_h, fused, b1, hbuf, FLD, FLD, 0, FLD);

  // 6) mlp2 GEMM (+bias) -> d_out [B,256,S] f32
  gemm_wmma<false, false><<<dim3(SS / 128, DD / 64, BB), 256, 0, stream>>>(
      w2_h, hbuf, b2, d_out, DD, FLD, 0, 0);
}
